// VectorQuantizer_34084860461508
// MI455X (gfx1250) — matrix operand order repaired and checked by reading, not compiled, not run on hardware
//
#include <hip/hip_runtime.h>
#include <hip/hip_bf16.h>
#include <stdint.h>

#define NUM_EMB   512
#define EMB_DIM   128
#define N_TOK     131072        // 32*64*64 tokens
#define QTOT      16777216      // N_TOK * EMB_DIM
#define HW        4096          // 64*64
#define BSTRIDE   524288        // EMB_DIM * HW floats per batch image
#define ROW_HALFS 136           // padded bf16 row stride (272B: 16B-aligned, bank-rotating)
#define ROW_BYTES (ROW_HALFS * 2)

// dynamic LDS layout for vq_argmin
#define ZS_OFF    0
#define ZS_BYTES  (8 * 16 * ROW_BYTES)        // 34,816  (8 waves x 16 tokens)
#define EB_OFF    ZS_BYTES
#define EB_BYTES  (NUM_EMB * ROW_BYTES)       // 139,264 (full bf16 codebook)
#define EN_OFF    (EB_OFF + EB_BYTES)         // 174,080
#define SMEM_TOTAL (EN_OFF + NUM_EMB * 4)     // 176,128 bytes (< 320KB WGP LDS)

typedef __attribute__((ext_vector_type(16))) __bf16 v16bf;
typedef __attribute__((ext_vector_type(8)))  float  v8f;
typedef __attribute__((ext_vector_type(4)))  float  fvec4;   // native vec for NT store

union FragBF { uint4 q[2]; v16bf v; };

__device__ __forceinline__ unsigned short f2bf(float f) {
    uint32_t u = __builtin_bit_cast(uint32_t, f);
    u = (u + 0x7FFFu + ((u >> 16) & 1u)) >> 16;   // round-to-nearest-even
    return (unsigned short)u;
}

// ---------------- Kernel A: codebook -> bf16, e-norms, zero loss ----------------
__global__ __launch_bounds__(EMB_DIM)
void vq_prep(const float* __restrict__ emb,
             unsigned short* __restrict__ emb_bf,
             float* __restrict__ e_norm,
             float* __restrict__ loss) {
    const int k = blockIdx.x, d = threadIdx.x;
    const float v = emb[k * EMB_DIM + d];
    emb_bf[k * EMB_DIM + d] = f2bf(v);
    float s = v * v;
    #pragma unroll
    for (int off = 16; off; off >>= 1) s += __shfl_xor(s, off, 32);
    __shared__ float part[4];
    if ((threadIdx.x & 31) == 0) part[threadIdx.x >> 5] = s;
    __syncthreads();
    if (threadIdx.x == 0) {
        e_norm[k] = part[0] + part[1] + part[2] + part[3];
        if (k == 0) *loss = 0.0f;
    }
}

// ---------------- Kernel B: WMMA distance GEMM + argmin ----------------
// 8 waves/workgroup; each wave owns 16 tokens vs all 512 codes.
// Codebook is async-copied global->LDS (ASYNCcnt) and shared by all waves;
// B fragments are double-buffered in VGPRs so ds_loads hide under the WMMAs.
__global__ __launch_bounds__(256, 1)
void vq_argmin(const float* __restrict__ z,
               const unsigned short* __restrict__ emb_bf,
               const float* __restrict__ e_norm,
               int* __restrict__ idx_out) {
    extern __shared__ char smem[];
    const int tid  = threadIdx.x;
    const int lane = tid & 31, wave = tid >> 5;
    const int tbase = blockIdx.x * 128 + wave * 16;   // 16 tokens per wave

    // ---- 1) async global->LDS copy of bf16 codebook + norms (CDNA5 path) ----
    for (int i = tid; i < (EB_BYTES >> 4); i += 256) {
        const int row = i >> 4, off = (i & 15) << 4;     // 16 x 16B chunks per row
        const unsigned ldsa = (unsigned)(uintptr_t)(smem + EB_OFF + row * ROW_BYTES + off);
        const char* g = (const char*)emb_bf + row * 256 + off;
        asm volatile("global_load_async_to_lds_b128 %0, %1, off"
                     :: "v"(ldsa), "v"(g) : "memory");
    }
    if (tid < 128) {
        const unsigned ldsa = (unsigned)(uintptr_t)(smem + EN_OFF + tid * 16);
        const char* g = (const char*)e_norm + tid * 16;
        asm volatile("global_load_async_to_lds_b128 %0, %1, off"
                     :: "v"(ldsa), "v"(g) : "memory");
    }

    // ---- 2) overlap: stage+transpose+bf16-convert this wave's 16 tokens ----
    {
        const int b = tbase >> 12, t0 = tbase & 4095;    // group never crosses batch
        const float* zb = z + (size_t)b * BSTRIDE + t0;
        const int tok = lane & 15, dh = lane >> 4;
        unsigned short* zp =
            (unsigned short*)(smem + ZS_OFF + (wave * 16 + tok) * ROW_BYTES);
        for (int j = 0; j < 64; ++j) {
            const int d = j * 2 + dh;
            zp[d] = f2bf(zb[(size_t)d * HW + tok]);
        }
    }
    asm volatile("s_wait_asynccnt 0x0" ::: "memory");
    __syncthreads();

    // ---- 3) A fragments (16x32 bf16 x 4 k-slices), documented wave32 layout ----
    v16bf a[4];
    {
        const char* base = smem + ZS_OFF + (wave * 16 + (lane & 15)) * ROW_BYTES
                         + (lane >> 4) * 16;
        #pragma unroll
        for (int ks = 0; ks < 4; ++ks) {
            FragBF f;
            f.q[0] = *(const uint4*)(base + ks * 64);
            f.q[1] = *(const uint4*)(base + ks * 64 + 32);
            a[ks] = f.v;
        }
    }

    const int col = lane & 15, khalf = lane >> 4;
    const char* ebase = smem + EB_OFF + col * ROW_BYTES + khalf * 16;
    const float* enb  = (const float*)(smem + EN_OFF);

    auto loadB = [&](FragBF* bf, int ct) {
        const char* p = ebase + ct * (16 * ROW_BYTES);
        #pragma unroll
        for (int ks = 0; ks < 4; ++ks) {
            bf[ks].q[0] = *(const uint4*)(p + ks * 64);
            bf[ks].q[1] = *(const uint4*)(p + ks * 64 + 32);
        }
    };

    float mv[8]; int mi[8];
    #pragma unroll
    for (int r = 0; r < 8; ++r) { mv[r] = 3.4e38f; mi[r] = 0; }

    // per-lane strict '<' keeps first occurrence (codes ascend with ct)
    auto upd = [&](const v8f& c, int code, float en) {
        #pragma unroll
        for (int r = 0; r < 8; ++r) {
            const float dist = __builtin_fmaf(-2.0f, c[r], en);  // ||e||^2 - 2 z.e
            const bool lt = dist < mv[r];
            mv[r] = lt ? dist : mv[r];
            mi[r] = lt ? code : mi[r];
        }
    };

    FragBF b0[4], b1[4];
    loadB(b0, 0);
    for (int ct = 0; ct < 32; ct += 2) {
        loadB(b1, ct + 1);                       // prefetch while WMMAs run on b0
        {
            v8f c = {};
            #pragma unroll
            for (int ks = 0; ks < 4; ++ks)
                c = __builtin_amdgcn_wmma_f32_16x16x32_bf16(
                        false, a[ks], false, b0[ks].v, (short)0, c, false, false);
            upd(c, ct * 16 + col, enb[ct * 16 + col]);
        }
        if (ct + 2 < 32) loadB(b0, ct + 2);      // prefetch while WMMAs run on b1
        {
            v8f c = {};
            #pragma unroll
            for (int ks = 0; ks < 4; ++ks)
                c = __builtin_amdgcn_wmma_f32_16x16x32_bf16(
                        false, a[ks], false, b1[ks].v, (short)0, c, false, false);
            upd(c, (ct + 1) * 16 + col, enb[(ct + 1) * 16 + col]);
        }
    }

    // ---- 4) cross-lane argmin over 16 lanes per token row (xor keeps halves) ----
    #pragma unroll
    for (int off = 1; off < 16; off <<= 1) {
        #pragma unroll
        for (int r = 0; r < 8; ++r) {
            const float ov = __shfl_xor(mv[r], off, 32);
            const int   oi = __shfl_xor(mi[r], off, 32);
            const bool take = (ov < mv[r]) || (ov == mv[r] && oi < mi[r]);
            mv[r] = take ? ov : mv[r];
            mi[r] = take ? oi : mi[r];
        }
    }
    if (col == 0) {
        const int base = tbase + khalf * 8;       // lanes hold rows 0-7 / 8-15
        #pragma unroll
        for (int r = 0; r < 8; ++r) idx_out[base + r] = mi[r];
    }
}

// ---------------- Kernel C: gather + straight-through + fused loss ----------------
__global__ __launch_bounds__(256)
void vq_gather(const float* __restrict__ z,
               const float* __restrict__ emb,
               const int* __restrict__ idx,
               float* __restrict__ out,
               float* __restrict__ loss) {
    const size_t base = ((size_t)blockIdx.x * 256 + threadIdx.x) * 4;
    const int tok = (int)(base >> 7);
    const int d   = (int)(base & 127);
    const int k   = idx[tok];
    const fvec4 zv = *(const fvec4*)(z + base);
    const fvec4 qv = *(const fvec4*)(emb + (size_t)k * EMB_DIM + d);
    const fvec4 df = qv - zv;                    // stop_grad(q - z)
    const fvec4 o  = zv + df;                    // straight-through estimator
    __builtin_nontemporal_store(o, (fvec4*)(out + base));   // write-once stream
    float s = df.x * df.x + df.y * df.y + df.z * df.z + df.w * df.w;
    #pragma unroll
    for (int off = 16; off; off >>= 1) s += __shfl_xor(s, off, 32);
    if ((threadIdx.x & 31) == 0)
        atomicAdd(loss, s * (1.25f / (float)QTOT));  // commitment(0.25) + embedding(1.0)
}

extern "C" void kernel_launch(void* const* d_in, const int* in_sizes, int n_in,
                              void* d_out, int out_size, void* d_ws, size_t ws_size,
                              hipStream_t stream) {
    const float* z   = (const float*)d_in[0];
    const float* emb = (const float*)d_in[1];
    float* out  = (float*)d_out;                         // quantized_st [32,128,64,64]
    int*   idxo = (int*)((float*)d_out + QTOT);          // indices [32,64,64] (int32)
    float* loss = (float*)d_out + QTOT + N_TOK;          // scalar loss

    unsigned short* emb_bf = (unsigned short*)d_ws;                       // 512*128 bf16
    float* e_norm = (float*)((char*)d_ws + NUM_EMB * EMB_DIM * 2);        // 512 f32

    (void)hipFuncSetAttribute((const void*)vq_argmin,
                              hipFuncAttributeMaxDynamicSharedMemorySize, SMEM_TOTAL);

    vq_prep  <<<NUM_EMB, EMB_DIM, 0, stream>>>(emb, emb_bf, e_norm, loss);
    vq_argmin<<<N_TOK / 128, 256, SMEM_TOTAL, stream>>>(z, emb_bf, e_norm, idxo);
    vq_gather<<<QTOT / 4 / 256, 256, 0, stream>>>(z, emb, idxo, out, loss);
}
